// SpectralCore_5411658793004
// MI455X (gfx1250) — matrix operand order repaired and checked by reading, not compiled, not run on hardware
//
#include <hip/hip_runtime.h>
#include <hip/hip_bf16.h>

// ---------------------------------------------------------------------------
// SpectralCore fused pipeline for MI455X (gfx1250, wave32, WMMA bf16)
//
// Math refactor (exact):
//   latf[t][e]  = sum_d latent[t][d] * enc_w[d][e]          (16x256 per head)
//   logits[t,n] = sum_e x[n][e]*latf[t][e]   (+lbias[t] cancels in softmax)
//   p = softmax_n(logits);   y[t] = sum_n p[t,n] * x[n]     (16x256, input space)
//   lt[t][d'] = sum_e (y[t][e]/S) * enc_w[256+d'][e] + enc_b[256+d'] + latent[t][d']
//   out = lt + sum_m sin(lt*pi*m/12)*W[:,m] + cos(..)*W[:,12+m]
// ---------------------------------------------------------------------------

#define Bn     16
#define Nn     4096
#define Cc     2048
#define Hh     8
#define Dd     256
#define Tt     16
#define NSPLIT 8
#define CHUNK  (Nn / NSPLIT)   // 512
#define TILE   32
#define ITERS  (CHUNK / TILE)  // 16

typedef __attribute__((ext_vector_type(16))) __bf16        v16bf;
typedef __attribute__((ext_vector_type(2)))  __bf16        v2bf;
typedef __attribute__((ext_vector_type(8)))  float         v8f;
typedef __attribute__((ext_vector_type(2)))  float         v2f;
typedef __attribute__((ext_vector_type(4)))  unsigned int  v4u;

union FragBF { v16bf bf; v4u u[2]; };

__device__ __forceinline__ unsigned short f2bf16(float x) {
  union { __bf16 h; unsigned short s; } u;
  u.h = (__bf16)x;   // native RNE converter
  return u.s;
}
// pair converter -> single v_cvt_pk_bf16_f32 vD, vA, vB (no mov_b16 shuffles)
__device__ __forceinline__ unsigned int pack2(float lo, float hi) {
  v2f f; f.x = lo; f.y = hi;
  union { v2bf b; unsigned int u; } u;
  u.b = __builtin_convertvector(f, v2bf);
  return u.u;
}

// A fragment (16xK tile, row-major LDS): lane holds row M = m0 + (lane&15).
// lanes 0-15 take K = k0+{0..15}; lanes 16-31 take K = k0+{16..31}.
__device__ __forceinline__ v16bf load_a_lds(const unsigned short* base, int stride,
                                            int m0, int k0, int ln, int hl) {
  FragBF f;
  const unsigned short* r = base + (m0 + ln) * stride;
  f.u[0] = *(const v4u*)(r + k0 + 16 * hl);
  f.u[1] = *(const v4u*)(r + k0 + 16 * hl + 8);
  return f.bf;
}
// B fragment (KxN tile) from an LDS array laid out [N][K]:
// lane holds column N = n0 + (lane&15); lanes 0-15 take K=k0..k0+15,
// lanes 16-31 take K=k0+16..k0+31 (2 packed per VGPR, memory order).
__device__ __forceinline__ v16bf load_b_lds(const unsigned short* base, int stride,
                                            int n0, int k0, int ln, int hl) {
  FragBF f;
  const unsigned short* r = base + (n0 + ln) * stride + k0 + 16 * hl;
  f.u[0] = *(const v4u*)(r);
  f.u[1] = *(const v4u*)(r + 8);
  return f.bf;
}

__device__ __forceinline__ v8f wmma_bf16(v16bf a, v16bf b, v8f c) {
  return __builtin_amdgcn_wmma_f32_16x16x32_bf16(false, a, false, b, (short)0, c,
                                                 false, false);
}

// ---------------------------------------------------------------------------
// Kernel 0: latf[h][t][e] = sum_d latent[h][t][d] * enc_w[d][e].
// Tiny (8 MMAC total) -- plain VALU, grid = H.
// ---------------------------------------------------------------------------
__global__ __launch_bounds__(256) void latf_kernel(
    const float* __restrict__ latent, const float* __restrict__ enc_w,
    float* __restrict__ latf) {
  const int h = blockIdx.x, tid = threadIdx.x;
  const float* lat = latent + (size_t)h * Tt * Dd;
  for (int idx = tid; idx < Tt * Dd; idx += 256) {
    const int t = idx >> 8, e = idx & 255;
    float s = 0.f;
    for (int d = 0; d < Dd; ++d) s += lat[t * Dd + d] * enc_w[d * Dd + e];
    latf[(size_t)h * Tt * Dd + idx] = s;
  }
}

// ---------------------------------------------------------------------------
// Kernel 1: flash-style partial attention over a chunk of N.
// grid = B*H*NSPLIT blocks, 128 threads (4 waves).
// Software-pipelined: tile it+1's global_load_b128s are issued before the
// compute barrier of tile it, so HBM latency hides behind WMMA/softmax.
// Emits: pm (running max), ps (running sumexp), py = sum_n exp(l-m) x[n] (16x256)
// ---------------------------------------------------------------------------
__global__ __launch_bounds__(128) void attn_partial_kernel(
    const float* __restrict__ emb, const float* __restrict__ latf,
    float* __restrict__ pm, float* __restrict__ ps, float* __restrict__ py) {
  __shared__ __align__(16) unsigned short s_latf[Tt * Dd];   // 8 KB
  __shared__ __align__(16) unsigned short s_xrow[TILE * Dd]; // 16 KB  [n][e]
  __shared__ __align__(16) unsigned short s_xT[Dd * TILE];   // 16 KB  [e][n]
  __shared__ __align__(16) unsigned short s_p[Tt * TILE];    // 1 KB   [t][n]
  __shared__ float s_logits[Tt * TILE];
  __shared__ float s_f[Tt], s_m[Tt], s_s[Tt];

  const int tid = threadIdx.x;
  const int lane = tid & 31, wave = tid >> 5;
  const int ln = lane & 15, hl = (lane >> 4) & 1;
  const int bh = blockIdx.x / NSPLIT, split = blockIdx.x % NSPLIT;
  const int b = bh / Hh, h = bh % Hh;

  for (int i = tid; i < Tt * Dd; i += 128) s_latf[i] = f2bf16(latf[h * Tt * Dd + i]);
  for (int i = tid; i < Tt * TILE; i += 128) s_logits[i] = 0.f;
  if (tid < Tt) { s_m[tid] = -1e30f; s_s[tid] = 0.f; }

  v8f acc[4] = {};  // this wave's 4 d-tiles of y (16x256 total across 4 waves)

  // loader mapping: each thread owns a row PAIR (prow, prow+1) x 32 columns,
  // so transposed LDS writes are packed b32 and row-major writes are b64/b128.
  const int prow = (tid >> 3) * 2;      // 0,2,...,30
  const int pseg = (tid & 7) * 32;      // 32-float segment within the row
  const float* gbase =
      emb + ((size_t)(b * Nn + split * CHUNK + prow)) * Cc + h * Dd + pseg;

  float4 stg0[8], stg1[8];
  {  // prologue: stage tile 0
    const float4* g0 = (const float4*)gbase;
    const float4* g1 = (const float4*)(gbase + Cc);
#pragma unroll
    for (int q = 0; q < 8; ++q) { stg0[q] = g0[q]; stg1[q] = g1[q]; }
  }

  for (int it = 0; it < ITERS; ++it) {
    // commit staged tile to LDS (bf16, both layouts)
#pragma unroll
    for (int q = 0; q < 8; ++q) {
      const int col = pseg + q * 4;
      uint2 r0, r1;
      r0.x = pack2(stg0[q].x, stg0[q].y); r0.y = pack2(stg0[q].z, stg0[q].w);
      r1.x = pack2(stg1[q].x, stg1[q].y); r1.y = pack2(stg1[q].z, stg1[q].w);
      *(uint2*)&s_xrow[prow * Dd + col]       = r0;
      *(uint2*)&s_xrow[(prow + 1) * Dd + col] = r1;
      *(unsigned int*)&s_xT[(col + 0) * TILE + prow] = pack2(stg0[q].x, stg1[q].x);
      *(unsigned int*)&s_xT[(col + 1) * TILE + prow] = pack2(stg0[q].y, stg1[q].y);
      *(unsigned int*)&s_xT[(col + 2) * TILE + prow] = pack2(stg0[q].z, stg1[q].z);
      *(unsigned int*)&s_xT[(col + 3) * TILE + prow] = pack2(stg0[q].w, stg1[q].w);
    }
    // issue next tile's loads now; they complete during this tile's compute
    if (it + 1 < ITERS) {
      const float* gn = gbase + (size_t)(it + 1) * TILE * Cc;
      const float4* g0 = (const float4*)gn;
      const float4* g1 = (const float4*)(gn + Cc);
#pragma unroll
      for (int q = 0; q < 8; ++q) { stg0[q] = g0[q]; stg1[q] = g1[q]; }
      __builtin_prefetch(gn + (size_t)TILE * Cc, 0, 1);  // prefetch tile it+2
    }
    __syncthreads();

    // logits(16x32) = latf(16x256) @ x_tile^T: all 4 waves busy.
    // wave -> (col-tile nt = wave>>1, K-half kh = wave&1); ds_add_f32 reduce.
    {
      const int nt = wave >> 1, kh = wave & 1;
      v8f c = {};
#pragma unroll
      for (int k0 = kh * 128; k0 < kh * 128 + 128; k0 += 32) {
        v16bf a  = load_a_lds(s_latf, Dd, 0, k0, ln, hl);
        v16bf bb = load_b_lds(s_xrow, Dd, nt * 16, k0, ln, hl);
        c = wmma_bf16(a, bb, c);
      }
#pragma unroll
      for (int r = 0; r < 8; ++r) {
        const int t = r + 8 * hl;
        atomicAdd(&s_logits[t * TILE + nt * 16 + ln], c[r]);
      }
    }
    __syncthreads();

    // online softmax per latent token (16 threads, 32 logits each);
    // re-zeroes s_logits for the next iteration's atomics.
    if (tid < Tt) {
      const int t = tid;
      float lrow[TILE], tm = -1e30f;
#pragma unroll
      for (int j = 0; j < TILE; ++j) {
        lrow[j] = s_logits[t * TILE + j];
        tm = fmaxf(tm, lrow[j]);
      }
#pragma unroll
      for (int j = 0; j < TILE; ++j) s_logits[t * TILE + j] = 0.f;
      const float mOld = s_m[t];
      const float nm = fmaxf(mOld, tm);
      const float fac = __expf(mOld - nm);
      float psum = 0.f;
#pragma unroll
      for (int j = 0; j < TILE; j += 2) {
        const float e0 = __expf(lrow[j] - nm);
        const float e1 = __expf(lrow[j + 1] - nm);
        psum += e0 + e1;
        *(unsigned int*)&s_p[t * TILE + j] = pack2(e0, e1);
      }
      s_s[t] = s_s[t] * fac + psum;
      s_m[t] = nm;
      s_f[t] = fac;
    }
    __syncthreads();

    // y(16x256) = rescale(y) + P(16x32) @ x_tile(32x256): 4 d-tiles per wave
    float fr[8];
#pragma unroll
    for (int r = 0; r < 8; ++r) fr[r] = s_f[r + 8 * hl];
    const v16bf pa = load_a_lds(s_p, TILE, 0, 0, ln, hl);
#pragma unroll
    for (int j = 0; j < 4; ++j) {
      const int ct = wave * 4 + j;
      const v16bf bb = load_b_lds(s_xT, TILE, ct * 16, 0, ln, hl);
#pragma unroll
      for (int r = 0; r < 8; ++r) acc[j][r] *= fr[r];
      acc[j] = wmma_bf16(pa, bb, acc[j]);
    }
    __syncthreads();
  }

  // write partials
  const size_t pbase = (size_t)(bh * NSPLIT + split) * Tt;
  if (tid < Tt) { pm[pbase + tid] = s_m[tid]; ps[pbase + tid] = s_s[tid]; }
#pragma unroll
  for (int j = 0; j < 4; ++j) {
    const int ct = wave * 4 + j;
#pragma unroll
    for (int r = 0; r < 8; ++r) {
      const int t = r + 8 * hl, d = ct * 16 + ln;
      py[(pbase + t) * Dd + d] = acc[j][r];
    }
  }
}

// ---------------------------------------------------------------------------
// Kernel 2: combine split partials, project with enc_w_v (WMMA), add bias +
// latent, apply the sin/cos spectral head, write out. grid = B*H, 128 threads.
// ---------------------------------------------------------------------------
__global__ __launch_bounds__(128) void finalize_kernel(
    const float* __restrict__ latent, const float* __restrict__ enc_w,
    const float* __restrict__ enc_b, const float* __restrict__ weights,
    const float* __restrict__ pm, const float* __restrict__ ps,
    const float* __restrict__ py, float* __restrict__ out) {
  __shared__ __align__(16) unsigned short s_ybar[Tt * Dd]; // normalized y, bf16
  __shared__ float s_wgt[Tt * NSPLIT];

  const int tid = threadIdx.x;
  const int lane = tid & 31, wave = tid >> 5;
  const int ln = lane & 15, hl = (lane >> 4) & 1;
  const int bh = blockIdx.x, b = bh / Hh, h = bh % Hh;
  const size_t pbase = (size_t)bh * NSPLIT * Tt;

  if (tid < Tt) {
    const int t = tid;
    float M = -1e30f;
#pragma unroll
    for (int c = 0; c < NSPLIT; ++c) M = fmaxf(M, pm[pbase + c * Tt + t]);
    float S = 0.f;
#pragma unroll
    for (int c = 0; c < NSPLIT; ++c)
      S += __expf(pm[pbase + c * Tt + t] - M) * ps[pbase + c * Tt + t];
    const float inv = 1.f / S;
#pragma unroll
    for (int c = 0; c < NSPLIT; ++c)
      s_wgt[t * NSPLIT + c] = __expf(pm[pbase + c * Tt + t] - M) * inv;
  }
  __syncthreads();

  for (int idx = tid; idx < Tt * Dd; idx += 128) {
    const int t = idx >> 8, e = idx & 255;
    float v = 0.f;
#pragma unroll
    for (int c = 0; c < NSPLIT; ++c)
      v += s_wgt[t * NSPLIT + c] * py[(pbase + c * Tt + t) * Dd + e];
    s_ybar[t * Dd + e] = f2bf16(v);
  }
  __syncthreads();

  // lt(16x256) = ybar(16x256) @ enc_w_v^T(256x256); B frags straight from global
#pragma unroll
  for (int j = 0; j < 4; ++j) {
    const int dt = wave * 4 + j;
    v8f acc = {};
#pragma unroll
    for (int k0 = 0; k0 < Dd; k0 += 32) {
      const v16bf a = load_a_lds(s_ybar, Dd, 0, k0, ln, hl);
      // B(K=e, N=d') = enc_w[(256+d')*256 + e]; contiguous 16 K per lane-half
      const float* wr =
          enc_w + ((size_t)(Dd + dt * 16 + ln)) * Dd + k0 + 16 * hl;
      const float4 w0 = *(const float4*)(wr);
      const float4 w1 = *(const float4*)(wr + 4);
      const float4 w2 = *(const float4*)(wr + 8);
      const float4 w3 = *(const float4*)(wr + 12);
      FragBF f;
      f.u[0] = (v4u){pack2(w0.x, w0.y), pack2(w0.z, w0.w),
                     pack2(w1.x, w1.y), pack2(w1.z, w1.w)};
      f.u[1] = (v4u){pack2(w2.x, w2.y), pack2(w2.z, w2.w),
                     pack2(w3.x, w3.y), pack2(w3.z, w3.w)};
      acc = wmma_bf16(a, f.bf, acc);
    }
    const int dv = dt * 16 + ln;
    const float bias = enc_b[Dd + dv];
    const int cidx = h * Dd + dv;
    float wb[24];
#pragma unroll
    for (int m = 0; m < 24; ++m) wb[m] = weights[(size_t)cidx * 24 + m];
#pragma unroll
    for (int r = 0; r < 8; ++r) {
      const int t = r + 8 * hl;
      const float val = acc[r] + bias + latent[((size_t)h * Tt + t) * Dd + dv];
      float spec = 0.f;
#pragma unroll
      for (int m = 0; m < 12; ++m) {
        const float arg = val * (3.14159265358979f * (float)m / 12.0f);
        spec += __sinf(arg) * wb[m] + __cosf(arg) * wb[12 + m];
      }
      out[((size_t)(b * Tt + t)) * Cc + cidx] = val + spec;
    }
  }
}

// ---------------------------------------------------------------------------
extern "C" void kernel_launch(void* const* d_in, const int* in_sizes, int n_in,
                              void* d_out, int out_size, void* d_ws,
                              size_t ws_size, hipStream_t stream) {
  (void)in_sizes; (void)n_in; (void)out_size; (void)ws_size;
  const float* emb     = (const float*)d_in[0];
  const float* latent  = (const float*)d_in[1];
  const float* weights = (const float*)d_in[2];
  const float* enc_w   = (const float*)d_in[3];
  const float* enc_b   = (const float*)d_in[4];
  float* out = (float*)d_out;
  float* ws  = (float*)d_ws;

  // workspace layout (floats)
  float* latf = ws;            // 8*16*256     = 32768
  float* pm   = ws + 32768;    // 128*8*16     = 16384
  float* psum = ws + 49152;    // 128*8*16     = 16384
  float* py   = ws + 65536;    // 128*8*16*256 = 4194304  (~16.25 MiB total)

  latf_kernel<<<Hh, 256, 0, stream>>>(latent, enc_w, latf);
  attn_partial_kernel<<<Bn * Hh * NSPLIT, 128, 0, stream>>>(emb, latf, pm,
                                                            psum, py);
  finalize_kernel<<<Bn * Hh, 128, 0, stream>>>(latent, enc_w, enc_b, weights,
                                               pm, psum, py, out);
}
